// Channel_Selective_Conv_28767690949130
// MI455X (gfx1250) — matrix operand order repaired and checked by reading, not compiled, not run on hardware
//
#include <hip/hip_runtime.h>
#include <hip/hip_bf16.h>

typedef __bf16 bf16_t;
typedef __bf16 bf16x2 __attribute__((ext_vector_type(2)));
typedef __bf16 bf16x8 __attribute__((ext_vector_type(8)));
typedef __bf16 v16bf  __attribute__((ext_vector_type(16)));
typedef float  v8f    __attribute__((ext_vector_type(8)));
typedef unsigned int u32x4 __attribute__((ext_vector_type(4)));
typedef int  i32x4 __attribute__((ext_vector_type(4)));
typedef int  i32x8 __attribute__((ext_vector_type(8)));

#define C_IN   256
#define MED    128
#define K_OUT  128
#define HW     6400           // 80*80
#define NB     16
#define PTILE  64             // pixels per workgroup
// LDS map: [0,64K) xls f32[64][256] (stage2/topk). Stage1 aliases inside it:
//   [0,4K)   bx   bf16[64][32]   (B fragments, [p][k])
//   [4K,20K) xstage f32 ping/pong 2 x [32][64] (TDM destination)
// [64K,80K) hlds bf16[64][128]
#define SMEM_BYTES (64*1024 + 16*1024)

// ---------------------------------------------------------------------------
// Convert fp32 weights to bf16 once (reused by all 1600 blocks; ~128KB stays
// hot in L2/WGP$).
// ---------------------------------------------------------------------------
__global__ void convert_weights(const float* __restrict__ w1,
                                const float* __restrict__ w2,
                                bf16_t* __restrict__ w1b,
                                bf16_t* __restrict__ w2b) {
  int i = blockIdx.x * blockDim.x + threadIdx.x;
  if (i < MED * C_IN) {
    w1b[i] = (bf16_t)w1[i];
    w2b[i] = (bf16_t)w2[i];
  }
}

// Issue a TDM 2-D tile load: 32 rows (channels, stride HW floats) x 64 floats
// into LDS at lds_off. Uniform scalars -> descriptor lives in SGPRs.
__device__ __forceinline__ void tdm_load_tile(const float* gsrc, unsigned lds_off) {
  unsigned long long ga = (unsigned long long)(const void*)gsrc;
  u32x4 g0;
  g0[0] = 1u;                                            // count=1, user mode
  g0[1] = lds_off;                                       // lds_addr (bytes)
  g0[2] = (unsigned)ga;                                  // global_addr[31:0]
  g0[3] = (unsigned)((ga >> 32) & 0x1FFFFFFu) | (2u << 30); // addr[56:32]|type=2
  i32x8 g1;
  g1[0] = (int)(2u << 16);          // data_size=2 (4B), no mask/pad/iterate
  g1[1] = (int)(64u << 16);         // tensor_dim0 = 64
  g1[2] = (int)(32u << 16);         // tensor_dim1 = 32
  g1[3] = (int)(64u << 16);         // tile_dim0 = 64
  g1[4] = 32;                       // tile_dim1 = 32, tile_dim2 = 0
  g1[5] = HW;                       // tensor_dim0_stride = 6400
  g1[6] = 0;
  g1[7] = 0;
  i32x4 z4 = {0, 0, 0, 0};
  i32x8 z8 = {0, 0, 0, 0, 0, 0, 0, 0};
  // 6-arg form (clang-23 / therock-10.0 HIP headers on this toolchain)
  __builtin_amdgcn_tensor_load_to_lds(g0, g1, z4, z4, z8, 0);
}

// ---------------------------------------------------------------------------
// Fused: GEMM1 (w1 * x -> h) -> GEMM2 (w2 * h -> x_) -> stable rank top-k
// over x_ (sigmoid is monotone) -> gather & product.
// One workgroup = 64 pixels of one batch image, 8 waves, wave32.
// ---------------------------------------------------------------------------
__global__ __launch_bounds__(256)
void fused_mlp_topk(const float* __restrict__ x,
                    const bf16_t* __restrict__ w1b,
                    const float* __restrict__ b1,
                    const bf16_t* __restrict__ w2b,
                    const float* __restrict__ b2,
                    float* __restrict__ out) {
  extern __shared__ char smem[];
  float*  xls    = (float*)smem;                  // [PTILE][C_IN] 64KB
  bf16_t* bx     = (bf16_t*)smem;                 // [PTILE][32]    4KB (alias)
  float*  xstage = (float*)(smem + 4096);         // 2 x [32][64]  16KB (alias)
  bf16_t* hlds   = (bf16_t*)(smem + 64 * 1024);   // [PTILE][MED]  16KB

  const int tid   = threadIdx.x;
  const int lane  = tid & 31;
  const int wv    = tid >> 5;                     // wave 0..7
  const int b     = blockIdx.x / (HW / PTILE);
  const int p0    = (blockIdx.x % (HW / PTILE)) * PTILE;
  const int l15   = lane & 15;
  const int lhalf = (lane < 16) ? 0 : 1;

  const float* xb = x + (size_t)b * C_IN * HW;
  const unsigned xs_off = (unsigned)(size_t)(const void*)xstage; // LDS byte addr

  // ======================= Stage 1: h = w1 * x + b1 ========================
  v8f acc[4];
  for (int nt = 0; nt < 4; ++nt)
    for (int e = 0; e < 8; ++e) acc[nt][e] = 0.0f;

  // Prefetch chunk 0 via the Tensor Data Mover.
  if (wv == 0) tdm_load_tile(xb + p0, xs_off);

  for (int ic = 0; ic < C_IN / 32; ++ic) {
    const int cur = ic & 1;
    __builtin_amdgcn_s_wait_tensorcnt(0);   // chunk ic landed (issuing wave)
    __syncthreads();                        // publish TDM data to all waves

    // Prefetch next chunk into the other buffer while we compute this one.
    if (wv == 0 && ic + 1 < C_IN / 32)
      tdm_load_tile(xb + (size_t)(ic + 1) * 32 * HW + p0,
                    xs_off + (unsigned)((ic + 1) & 1) * 8192);

    // Transpose+convert xstage[cur] ([k][p] f32) -> bx ([p][k] bf16),
    // packed bf16x2 stores (k-pairs are adjacent in bx).
    {
      const float* xs = xstage + cur * (32 * 64);
      const int kp = tid >> 4;              // k-pair 0..15
      const int pp = (tid & 15) << 2;       // pixel 0..60
      float4 a4 = *(const float4*)(xs + (2 * kp + 0) * 64 + pp);
      float4 b4 = *(const float4*)(xs + (2 * kp + 1) * 64 + pp);
      bf16x2 t;
      t[0] = (bf16_t)a4.x; t[1] = (bf16_t)b4.x; *(bf16x2*)(bx + (pp + 0) * 32 + 2 * kp) = t;
      t[0] = (bf16_t)a4.y; t[1] = (bf16_t)b4.y; *(bf16x2*)(bx + (pp + 1) * 32 + 2 * kp) = t;
      t[0] = (bf16_t)a4.z; t[1] = (bf16_t)b4.z; *(bf16x2*)(bx + (pp + 2) * 32 + 2 * kp) = t;
      t[0] = (bf16_t)a4.w; t[1] = (bf16_t)b4.w; *(bf16x2*)(bx + (pp + 3) * 32 + 2 * kp) = t;
    }
    __syncthreads();

    // A fragment (16x32 bf16): lane m = l15, K halves {0..7,16..23}/{8..15,24..31}
    const int kc = ic * 32;
    const bf16_t* ap = w1b + (size_t)(wv * 16 + l15) * C_IN + kc + lhalf * 8;
    bf16x8 alo = *(const bf16x8*)ap;
    bf16x8 ahi = *(const bf16x8*)(ap + 16);
    v16bf a = __builtin_shufflevector(alo, ahi, 0,1,2,3,4,5,6,7,8,9,10,11,12,13,14,15);

    for (int nt = 0; nt < 4; ++nt) {
      const bf16_t* bp = bx + (nt * 16 + l15) * 32 + lhalf * 8;
      bf16x8 blo = *(const bf16x8*)bp;
      bf16x8 bhi = *(const bf16x8*)(bp + 16);
      v16bf bm = __builtin_shufflevector(blo, bhi, 0,1,2,3,4,5,6,7,8,9,10,11,12,13,14,15);
      acc[nt] = __builtin_amdgcn_wmma_f32_16x16x32_bf16(
          false, a, false, bm, (short)0, acc[nt], false, false);
    }
    __syncthreads();   // bx/xstage[cur] consumed before next overwrite
  }

  // bias b1, convert to bf16, store h tile to LDS [p][m] as one b128 per tile
  {
    const int mbase = wv * 16 + lhalf * 8;
    float b1r[8];
    for (int r = 0; r < 8; ++r) b1r[r] = b1[mbase + r];
    for (int nt = 0; nt < 4; ++nt) {
      int p = nt * 16 + l15;
      bf16x8 hv;
      for (int r = 0; r < 8; ++r) hv[r] = (bf16_t)(acc[nt][r] + b1r[r]);
      *(bf16x8*)(hlds + p * MED + mbase) = hv;
    }
  }
  __syncthreads();

  // ======================= Stage 2: x_ = w2 * h + b2 =======================
  v8f acc2[2][4];
  for (int s = 0; s < 2; ++s)
    for (int nt = 0; nt < 4; ++nt)
      for (int e = 0; e < 8; ++e) acc2[s][nt][e] = 0.0f;

  for (int kc = 0; kc < MED; kc += 32) {
    v16bf a2[2];
    for (int s = 0; s < 2; ++s) {
      const bf16_t* ap = w2b + (size_t)(s * 128 + wv * 16 + l15) * MED + kc + lhalf * 8;
      bf16x8 alo = *(const bf16x8*)ap;
      bf16x8 ahi = *(const bf16x8*)(ap + 16);
      a2[s] = __builtin_shufflevector(alo, ahi, 0,1,2,3,4,5,6,7,8,9,10,11,12,13,14,15);
    }
    for (int nt = 0; nt < 4; ++nt) {
      const bf16_t* bp = hlds + (nt * 16 + l15) * MED + kc + lhalf * 8;
      bf16x8 blo = *(const bf16x8*)bp;
      bf16x8 bhi = *(const bf16x8*)(bp + 16);
      v16bf bm = __builtin_shufflevector(blo, bhi, 0,1,2,3,4,5,6,7,8,9,10,11,12,13,14,15);
      acc2[0][nt] = __builtin_amdgcn_wmma_f32_16x16x32_bf16(
          false, a2[0], false, bm, (short)0, acc2[0][nt], false, false);
      acc2[1][nt] = __builtin_amdgcn_wmma_f32_16x16x32_bf16(
          false, a2[1], false, bm, (short)0, acc2[1][nt], false, false);
    }
  }

  // bias b2, write logits x_ to LDS [p][c] as float4 b128 stores
  for (int s = 0; s < 2; ++s) {
    const int cb = s * 128 + wv * 16 + lhalf * 8;
    float b2r[8];
    for (int r = 0; r < 8; ++r) b2r[r] = b2[cb + r];
    for (int nt = 0; nt < 4; ++nt) {
      int p = nt * 16 + l15;
      float4 lo = make_float4(acc2[s][nt][0] + b2r[0], acc2[s][nt][1] + b2r[1],
                              acc2[s][nt][2] + b2r[2], acc2[s][nt][3] + b2r[3]);
      float4 hi = make_float4(acc2[s][nt][4] + b2r[4], acc2[s][nt][5] + b2r[5],
                              acc2[s][nt][6] + b2r[6], acc2[s][nt][7] + b2r[7]);
      *(float4*)(xls + p * C_IN + cb)     = lo;
      *(float4*)(xls + p * C_IN + cb + 4) = hi;
    }
  }
  __syncthreads();

  // ====================== Stage 3: stable rank top-k =======================
  // rank(i) = #{j: v_j > v_i} + #{j < i: v_j == v_i} reproduces
  // jax.lax.top_k ordering (incl. ties); ranks 0..127 fully cover the output.
  {
    const int p  = tid >> 2;          // 4 threads per pixel
    const int cg = (tid & 3) * 64;    // 64 channels each
    const float* row = xls + p * C_IN;
    const int hw = p0 + p;
    for (int ii = 0; ii < 64; ++ii) {
      int   i = cg + ii;
      float v = row[i];
      int rank = 0;
      for (int j4 = 0; j4 < C_IN; j4 += 4) {
        float4 q = *(const float4*)(row + j4);
        rank += (q.x > v) || (q.x == v && (j4 + 0) < i);
        rank += (q.y > v) || (q.y == v && (j4 + 1) < i);
        rank += (q.z > v) || (q.z == v && (j4 + 2) < i);
        rank += (q.w > v) || (q.w == v && (j4 + 3) < i);
      }
      if (rank < K_OUT) {
        float xv = xb[(size_t)i * HW + hw];                  // out1 gather
        out[((size_t)b * K_OUT + rank) * HW + hw] = xv * v;  // out1*out2
      }
    }
  }
}

extern "C" void kernel_launch(void* const* d_in, const int* in_sizes, int n_in,
                              void* d_out, int out_size, void* d_ws, size_t ws_size,
                              hipStream_t stream) {
  (void)in_sizes; (void)n_in; (void)out_size; (void)ws_size;
  const float* x  = (const float*)d_in[0];
  const float* w1 = (const float*)d_in[1];
  const float* b1 = (const float*)d_in[2];
  const float* w2 = (const float*)d_in[3];
  const float* b2 = (const float*)d_in[4];
  float* out = (float*)d_out;

  bf16_t* w1b = (bf16_t*)d_ws;                 // 128*256 bf16 = 64KB
  bf16_t* w2b = w1b + MED * C_IN;              // 256*128 bf16 = 64KB

  convert_weights<<<dim3((MED * C_IN + 255) / 256), dim3(256), 0, stream>>>(
      w1, w2, w1b, w2b);

  (void)hipFuncSetAttribute((const void*)fused_mlp_topk,
                            hipFuncAttributeMaxDynamicSharedMemorySize,
                            SMEM_BYTES);
  dim3 grid(NB * (HW / PTILE));                // 16 * 100 = 1600 workgroups
  fused_mlp_topk<<<grid, dim3(256), SMEM_BYTES, stream>>>(
      x, w1b, b1, w2b, b2, out);
}
